// EasyExprGNN_74071005987083
// MI455X (gfx1250) — matrix operand order repaired and checked by reading, not compiled, not run on hardware
//
#include <hip/hip_runtime.h>
#include <hip/hip_bf16.h>

// ---------------------------------------------------------------------------
// EasyExprGNN fused kernel for MI455X (gfx1250, wave32, WMMA).
//
// Algebraic restructure (exact, up to fp reassociation):
//   nagg = (sum_n relu(LN(x_n @ nW1 + nb1))) @ nW2 + N*nb2
//   x@hW1+hb1 = erelu@(sW2@hW1[0:32]) + hsum@(nW2@hW1[32:64]) + bias'
// Per-neighbor LN variance is a scalar quadratic form in (x0,x1) with
// precomputed channel covariances -> no per-neighbor reductions at all.
//
// Pre-pass (1 tiny block): combined 32x64 f16 weights (transposed for
// B-operand layout), folded bias, centered*gamma weight vectors, covariance
// scalars.  Main kernel: 1 wave = 16 rows, 12x v_wmma_f32_16x16x32_f16.
// ---------------------------------------------------------------------------

typedef __attribute__((ext_vector_type(16))) _Float16 v16h;
typedef __attribute__((ext_vector_type(8)))  float    v8f;

#define WS_WCS   0          // _Float16[64][32] : (sW2 @ hW1_top)^T   [n][k]
#define WS_WCN   4096       // _Float16[64][32] : (nW2 @ hW1_bot)^T   [n][k]
#define WS_W2T   8192       // _Float16[32][64] : hW2^T               [n][k]
#define WS_BIASC 12288      // float[64] : hb1 + sb2@hW1_top + 32*nb2@hW1_bot
#define WS_NWG0  12544      // float[32] : (nW1[0]-mean)*nlg
#define WS_NWG1  12672      // float[32] : (nW1[1]-mean)*nlg
#define WS_NWGB  12800      // float[32] : (nb1 -mean)*nlg
#define WS_NBET  12928      // float[32] : nlb
#define WS_SWG   13056      // float[32] : (sW1-mean)*slg
#define WS_SWGB  13184      // float[32] : (sb1-mean)*slg
#define WS_SBET  13312      // float[32] : slb
#define WS_SCAL  13440      // float[16] : V00 V01 V11 V0b V1b Vbb T00 T0b Tbb

#define LN_EPS 1e-5f

__global__ __launch_bounds__(64)
void gnn_prepass(const float* __restrict__ nW1, const float* __restrict__ nb1,
                 const float* __restrict__ nlg, const float* __restrict__ nlb,
                 const float* __restrict__ nW2, const float* __restrict__ nb2,
                 const float* __restrict__ sW1, const float* __restrict__ sb1,
                 const float* __restrict__ slg, const float* __restrict__ slb,
                 const float* __restrict__ sW2, const float* __restrict__ sb2,
                 const float* __restrict__ hW1, const float* __restrict__ hb1,
                 const float* __restrict__ hW2, char* __restrict__ ws) {
  _Float16* WcS  = (_Float16*)(ws + WS_WCS);
  _Float16* WcN  = (_Float16*)(ws + WS_WCN);
  _Float16* W2t  = (_Float16*)(ws + WS_W2T);
  float* biasC   = (float*)(ws + WS_BIASC);
  float* nwg0    = (float*)(ws + WS_NWG0);
  float* nwg1    = (float*)(ws + WS_NWG1);
  float* nwgb    = (float*)(ws + WS_NWGB);
  float* nbet    = (float*)(ws + WS_NBET);
  float* swg     = (float*)(ws + WS_SWG);
  float* swgb    = (float*)(ws + WS_SWGB);
  float* sbet    = (float*)(ws + WS_SBET);
  float* scal    = (float*)(ws + WS_SCAL);

  const int tid = threadIdx.x;   // 64 threads
  if (tid < 64) {
    // column tid of the combined 32x64 matrices, stored transposed [n][k]
    for (int k = 0; k < 32; k++) {
      float a = 0.f, c = 0.f;
      for (int j = 0; j < 32; j++) {
        a += sW2[k * 32 + j] * hW1[j * 64 + tid];
        c += nW2[k * 32 + j] * hW1[(32 + j) * 64 + tid];
      }
      WcS[tid * 32 + k] = (_Float16)a;
      WcN[tid * 32 + k] = (_Float16)c;
    }
    float bc = hb1[tid];
    for (int j = 0; j < 32; j++) {
      bc += sb2[j] * hW1[j * 64 + tid];
      bc += 32.0f * nb2[j] * hW1[(32 + j) * 64 + tid];
    }
    biasC[tid] = bc;
  }
  if (tid < 32) {
    for (int kk = 0; kk < 64; kk++)
      W2t[tid * 64 + kk] = (_Float16)hW2[kk * 32 + tid];   // hW2 is (64,32)

    float mw0 = 0.f, mw1 = 0.f, mb = 0.f, msw = 0.f, msb = 0.f;
    for (int i = 0; i < 32; i++) {
      mw0 += nW1[i]; mw1 += nW1[32 + i]; mb += nb1[i];
      msw += sW1[i]; msb += sb1[i];
    }
    mw0 *= (1.f/32.f); mw1 *= (1.f/32.f); mb *= (1.f/32.f);
    msw *= (1.f/32.f); msb *= (1.f/32.f);
    nwg0[tid] = (nW1[tid]      - mw0) * nlg[tid];
    nwg1[tid] = (nW1[32 + tid] - mw1) * nlg[tid];
    nwgb[tid] = (nb1[tid]      - mb ) * nlg[tid];
    nbet[tid] = nlb[tid];
    swg [tid] = (sW1[tid] - msw) * slg[tid];
    swgb[tid] = (sb1[tid] - msb) * slg[tid];
    sbet[tid] = slb[tid];
  }
  if (tid == 0) {
    float mw0 = 0.f, mw1 = 0.f, mb = 0.f, msw = 0.f, msb = 0.f;
    for (int i = 0; i < 32; i++) {
      mw0 += nW1[i]; mw1 += nW1[32 + i]; mb += nb1[i];
      msw += sW1[i]; msb += sb1[i];
    }
    mw0 *= (1.f/32.f); mw1 *= (1.f/32.f); mb *= (1.f/32.f);
    msw *= (1.f/32.f); msb *= (1.f/32.f);
    float s00=0,s01=0,s11=0,s0b=0,s1b=0,sbb=0,t00=0,t0b=0,tbb=0;
    for (int i = 0; i < 32; i++) {
      float c0 = nW1[i] - mw0, c1 = nW1[32 + i] - mw1, cb = nb1[i] - mb;
      s00 += c0*c0; s01 += c0*c1; s11 += c1*c1;
      s0b += c0*cb; s1b += c1*cb; sbb += cb*cb;
      float cw = sW1[i] - msw, cq = sb1[i] - msb;
      t00 += cw*cw; t0b += cw*cq; tbb += cq*cq;
    }
    const float inv = 1.f/32.f;
    scal[0]=s00*inv; scal[1]=s01*inv; scal[2]=s11*inv;
    scal[3]=s0b*inv; scal[4]=s1b*inv; scal[5]=sbb*inv;
    scal[6]=t00*inv; scal[7]=t0b*inv; scal[8]=tbb*inv;
  }
}

__device__ __forceinline__ void load8f(float* d, const float* __restrict__ p) {
  float4 a = *(const float4*)p;
  float4 b = *(const float4*)(p + 4);
  d[0]=a.x; d[1]=a.y; d[2]=a.z; d[3]=a.w;
  d[4]=b.x; d[5]=b.y; d[6]=b.z; d[7]=b.w;
}

// B=8, S=4096, N=32, H=32, D=64, O=32.  Rows = B*S = 32768.
// Block: 256 threads = 8 waves; wave owns 16 rows. Grid: 256 blocks.
__global__ __launch_bounds__(256)
void gnn_main(const float* __restrict__ expr, const float* __restrict__ neigh,
              const float* __restrict__ edge, const int* __restrict__ mask,
              const float* __restrict__ hlg, const float* __restrict__ hlb,
              const float* __restrict__ hb2, const char* __restrict__ ws,
              float* __restrict__ out) {
  __shared__ float stage[8][16 * 64];   // per-wave 16x64 f32 transpose buffer

  const _Float16* WcS = (const _Float16*)(ws + WS_WCS);
  const _Float16* WcN = (const _Float16*)(ws + WS_WCN);
  const _Float16* W2t = (const _Float16*)(ws + WS_W2T);
  const float* biasC  = (const float*)(ws + WS_BIASC);
  const float* scal   = (const float*)(ws + WS_SCAL);

  const int lane = threadIdx.x & 31;
  const int wave = threadIdx.x >> 5;
  const int lo = lane & 15;            // row within tile / column within C tile
  const int hi = lane >> 4;            // lane-pair half: selects K/channel half
  const int tileBase = blockIdx.x * 128 + wave * 16;
  const int g = tileBase + lo;         // this lane-pair's (b,s) row
  const int b = g >> 12;
  const int s = g & 4095;

  // A-layout channel runs for this lane: {base0..+8} U {base1..+8}
  const int base0 = hi * 16;
  const int base1 = hi * 16 + 8;

  // ---- neighbor branch: 32 neighbors of Linear(2,32)+LN+ReLU, summed ----
  float wg0[16], wg1[16], wgb[16], bet[16];
  load8f(wg0,     (const float*)(ws + WS_NWG0) + base0);
  load8f(wg0 + 8, (const float*)(ws + WS_NWG0) + base1);
  load8f(wg1,     (const float*)(ws + WS_NWG1) + base0);
  load8f(wg1 + 8, (const float*)(ws + WS_NWG1) + base1);
  load8f(wgb,     (const float*)(ws + WS_NWGB) + base0);
  load8f(wgb + 8, (const float*)(ws + WS_NWGB) + base1);
  load8f(bet,     (const float*)(ws + WS_NBET) + base0);
  load8f(bet + 8, (const float*)(ws + WS_NBET) + base1);
  const float V00 = scal[0], V01 = scal[1], V11 = scal[2];
  const float V0b = scal[3], V1b = scal[4], Vbb = scal[5];

  float hsum[16];
  #pragma unroll
  for (int e = 0; e < 16; e++) hsum[e] = 0.f;

  const float* nb_base = neigh + (b * 32) * 4096 + s;   // + n*4096
  const float* eg_base = edge + (b * 4096 + s) * 32;    // + n
  for (int n = 0; n < 32; n++) {
    const float x0 = nb_base[n * 4096];
    const float x1 = eg_base[n];
    const float var = x0*x0*V00 + x1*x1*V11 + Vbb
                    + 2.f * (x0*x1*V01 + x0*V0b + x1*V1b);
    const float rstd = rsqrtf(var + LN_EPS);
    #pragma unroll
    for (int e = 0; e < 16; e++) {
      float t = fmaf(x0, wg0[e], fmaf(x1, wg1[e], wgb[e]));
      float v = fmaf(t, rstd, bet[e]);
      hsum[e] += fmaxf(v, 0.f);
    }
  }

  // ---- self branch: Linear(1,32)+LN+ReLU ----
  float sg[16], sgb[16], sbe[16], er[16];
  load8f(sg,      (const float*)(ws + WS_SWG) + base0);
  load8f(sg + 8,  (const float*)(ws + WS_SWG) + base1);
  load8f(sgb,     (const float*)(ws + WS_SWGB) + base0);
  load8f(sgb + 8, (const float*)(ws + WS_SWGB) + base1);
  load8f(sbe,     (const float*)(ws + WS_SBET) + base0);
  load8f(sbe + 8, (const float*)(ws + WS_SBET) + base1);
  {
    const float x = expr[g];
    const float var = x*x*scal[6] + 2.f*x*scal[7] + scal[8];
    const float rstd = rsqrtf(var + LN_EPS);
    #pragma unroll
    for (int e = 0; e < 16; e++) {
      float t = fmaf(x, sg[e], sgb[e]);
      er[e] = fmaxf(fmaf(t, rstd, sbe[e]), 0.f);
    }
  }

  // ---- WMMA stage 1: y1(16x64) = erelu@WcS + hsum@WcN + bias' ----
  v16h aE, aH;
  #pragma unroll
  for (int e = 0; e < 16; e++) { aE[e] = (_Float16)er[e]; aH[e] = (_Float16)hsum[e]; }

  #pragma unroll
  for (int t = 0; t < 4; t++) {
    const v16h bS = *(const v16h*)(WcS + (t * 16 + lo) * 32 + hi * 16);
    const v16h bN = *(const v16h*)(WcN + (t * 16 + lo) * 32 + hi * 16);
    v8f z = {};
    v8f acc = __builtin_amdgcn_wmma_f32_16x16x32_f16(false, aE, false, bS,
                                                     (short)0, z, false, false);
    acc = __builtin_amdgcn_wmma_f32_16x16x32_f16(false, aH, false, bN,
                                                 (short)0, acc, false, false);
    const float bc = biasC[t * 16 + lo];
    #pragma unroll
    for (int r = 0; r < 8; r++) {
      // C layout: value (m = r + 8*hi, n = t*16 + lo)
      stage[wave][(r + 8 * hi) * 64 + t * 16 + lo] = acc[r] + bc;
    }
  }
  __syncthreads();

  // ---- LN(64) + ReLU, re-read in A layout (transpose via LDS) ----
  float y[32];
  const float* srow = &stage[wave][lo * 64];
  #pragma unroll
  for (int c = 0; c < 2; c++)
    #pragma unroll
    for (int grp = 0; grp < 2; grp++)
      load8f(&y[c * 16 + grp * 8], srow + c * 32 + hi * 16 + grp * 8);

  float sum = 0.f, sq = 0.f;
  #pragma unroll
  for (int i = 0; i < 32; i++) { sum += y[i]; sq += y[i] * y[i]; }
  sum += __shfl_xor(sum, 16, 32);
  sq  += __shfl_xor(sq, 16, 32);
  const float mean = sum * (1.f / 64.f);
  const float var  = sq * (1.f / 64.f) - mean * mean;
  const float rstd = rsqrtf(var + LN_EPS);

  float hg[32], hb[32];
  #pragma unroll
  for (int c = 0; c < 2; c++)
    #pragma unroll
    for (int grp = 0; grp < 2; grp++) {
      const int start = c * 32 + hi * 16 + grp * 8;
      load8f(&hg[c * 16 + grp * 8], hlg + start);
      load8f(&hb[c * 16 + grp * 8], hlb + start);
    }

  v16h a0, a1;
  #pragma unroll
  for (int grp = 0; grp < 2; grp++)
    #pragma unroll
    for (int i = 0; i < 8; i++) {
      const int e = grp * 8 + i;
      float v0 = fmaf((y[e]      - mean) * rstd, hg[e],      hb[e]);
      float v1 = fmaf((y[16 + e] - mean) * rstd, hg[16 + e], hb[16 + e]);
      a0[e] = (_Float16)fmaxf(v0, 0.f);
      a1[e] = (_Float16)fmaxf(v1, 0.f);
    }

  // ---- WMMA stage 2: out(16x32) = a(16x64) @ hW2 + hb2, masked ----
  int mrow[8];
  #pragma unroll
  for (int r = 0; r < 8; r++) mrow[r] = mask[tileBase + r + 8 * hi];

  #pragma unroll
  for (int t2 = 0; t2 < 2; t2++) {
    const v16h b0 = *(const v16h*)(W2t + (t2 * 16 + lo) * 64 + hi * 16);
    const v16h b1 = *(const v16h*)(W2t + (t2 * 16 + lo) * 64 + 32 + hi * 16);
    v8f z = {};
    v8f d = __builtin_amdgcn_wmma_f32_16x16x32_f16(false, a0, false, b0,
                                                   (short)0, z, false, false);
    d = __builtin_amdgcn_wmma_f32_16x16x32_f16(false, a1, false, b1,
                                               (short)0, d, false, false);
    const float bias2 = hb2[t2 * 16 + lo];
    #pragma unroll
    for (int r = 0; r < 8; r++) {
      const int gr = tileBase + r + 8 * hi;
      const float val = mrow[r] ? 0.f : (d[r] + bias2);
      out[gr * 32 + t2 * 16 + lo] = val;
    }
  }
}

extern "C" void kernel_launch(void* const* d_in, const int* in_sizes, int n_in,
                              void* d_out, int out_size, void* d_ws, size_t ws_size,
                              hipStream_t stream) {
  const float* expr  = (const float*)d_in[0];
  const float* neigh = (const float*)d_in[1];
  const float* edge  = (const float*)d_in[2];
  const int*   mask  = (const int*)d_in[3];
  const float* nW1 = (const float*)d_in[4];
  const float* nb1 = (const float*)d_in[5];
  const float* nlg = (const float*)d_in[6];
  const float* nlb = (const float*)d_in[7];
  const float* nW2 = (const float*)d_in[8];
  const float* nb2 = (const float*)d_in[9];
  const float* sW1 = (const float*)d_in[10];
  const float* sb1 = (const float*)d_in[11];
  const float* slg = (const float*)d_in[12];
  const float* slb = (const float*)d_in[13];
  const float* sW2 = (const float*)d_in[14];
  const float* sb2 = (const float*)d_in[15];
  const float* hW1 = (const float*)d_in[16];
  const float* hb1 = (const float*)d_in[17];
  const float* hlg = (const float*)d_in[18];
  const float* hlb = (const float*)d_in[19];
  const float* hW2 = (const float*)d_in[20];
  const float* hb2 = (const float*)d_in[21];
  char* ws = (char*)d_ws;

  gnn_prepass<<<1, 64, 0, stream>>>(nW1, nb1, nlg, nlb, nW2, nb2,
                                    sW1, sb1, slg, slb, sW2, sb2,
                                    hW1, hb1, hW2, ws);
  gnn_main<<<256, 256, 0, stream>>>(expr, neigh, edge, mask,
                                    hlg, hlb, hb2, ws, (float*)d_out);
}
